// TransformerDecoder_16698832847175
// MI455X (gfx1250) — compile-verified, matrix operand order checked by reading, not run on hardware
//
#include <hip/hip_runtime.h>

// ---------------- types ----------------
typedef __attribute__((ext_vector_type(16))) _Float16 v16h;
typedef __attribute__((ext_vector_type(8)))  _Float16 v8h;
typedef __attribute__((ext_vector_type(8)))  float    v8f;

#define B_SZ   64
#define L_SZ   256
#define DIN_SZ 64
#define H_SZ   512
#define NH_SZ  16
#define DH_SZ  32
#define M_ROWS (B_SZ * L_SZ)          // 16384
#define MASK_BIG 100000000.0f

// Build a 16-half WMMA fragment from two contiguous 8-half (16 B) runs.
// Per ISA 16-bit layout: elements 0..7 are K = half*8..+7, elements 8..15 are
// K = 16+half*8..+7 -> two b128 loads when storage is K-contiguous.
static __device__ inline v16h load_frag(const _Float16* p0, const _Float16* p1)
{
    union { v16h f; v8h h[2]; } u;
    u.h[0] = *(const v8h*)p0;
    u.h[1] = *(const v8h*)p1;
    return u.f;
}

// ============================================================
// Kernel 1: per-batch mean over L, concat -> xcat_h (f16) [B,L,128]
// ============================================================
__global__ void __launch_bounds__(256)
mean_concat_kernel(const float* __restrict__ in, _Float16* __restrict__ xcat)
{
    __shared__ float part[256];
    __shared__ float us[DIN_SZ];
    const int t = threadIdx.x;
    const int b = blockIdx.x;
    const int d = t & 63, seg = t >> 6;
    float s = 0.f;
    for (int l = seg; l < L_SZ; l += 4)
        s += in[((size_t)b * L_SZ + l) * DIN_SZ + d];
    part[t] = s;
    __syncthreads();
    if (seg == 0)
        us[d] = (part[d] + part[64 + d] + part[128 + d] + part[192 + d]) * (1.f / L_SZ);
    __syncthreads();
    for (int idx = t; idx < L_SZ * 2 * DIN_SZ; idx += 256) {
        int l = idx >> 7, c = idx & 127;
        float v = (c < DIN_SZ) ? in[((size_t)b * L_SZ + l) * DIN_SZ + c] : us[c - DIN_SZ];
        xcat[((size_t)b * L_SZ + l) * (2 * DIN_SZ) + c] = (_Float16)v;
    }
}

// ============================================================
// Kernel 2a: weight convert  W[K,N] f32 -> Wc[N,K] f16 (column-major)
// ============================================================
__global__ void __launch_bounds__(256)
convert_w_kernel(const float* __restrict__ W, _Float16* __restrict__ Wc,
                 int K, int N)
{
    const int idx = blockIdx.x * 256 + threadIdx.x;   // grid covers K*N
    const int n = idx / K, k = idx - n * K;           // contiguous k -> contiguous write
    Wc[(size_t)n * K + k] = (_Float16)W[(size_t)k * N + n];
}

// ============================================================
// Kernel 2b: WMMA GEMM  C[M,N] = act(Ah[M,K] @ Wc[N,K]^T + bias)
// Both operands f16, K-contiguous -> every fragment is two
// global_load_b128 from L2-resident data. No LDS, no barriers.
// 256 threads = 8 waves; block tile 64x128; wave tile 32x32.
// cht: 0 = Ch row-major [M,N]; 1 = Ch transposed [B,NH,DH,L] (for V).
// ============================================================
__global__ void __launch_bounds__(256)
wmma_gemm_kernel(const _Float16* __restrict__ Ah, const _Float16* __restrict__ Wc,
                 const float* __restrict__ bias,
                 float* __restrict__ Cf, _Float16* __restrict__ Ch,
                 int M, int N, int K, int relu, int cht)
{
    const int lane = threadIdx.x & 31;
    const int wave = threadIdx.x >> 5;
    const int wm = wave >> 2;            // 0..1 : 32-row group
    const int wn = wave & 3;             // 0..3 : 32-col group
    const int m0 = blockIdx.y * 64;
    const int n0 = blockIdx.x * 128;
    const int mloc = lane & 15;
    const int half = lane >> 4;

    const _Float16* ar0 = Ah + (size_t)(m0 + wm * 32 + mloc) * K + half * 8;
    const _Float16* ar1 = ar0 + (size_t)16 * K;
    const _Float16* bc0 = Wc + (size_t)(n0 + wn * 32 + mloc) * K + half * 8;
    const _Float16* bc1 = bc0 + (size_t)16 * K;

    v8f acc[2][2] = {{{}, {}}, {{}, {}}};

    for (int k0 = 0; k0 < K; k0 += 32) {
        const v16h a0 = load_frag(ar0 + k0, ar0 + k0 + 16);
        const v16h a1 = load_frag(ar1 + k0, ar1 + k0 + 16);
        const v16h b0 = load_frag(bc0 + k0, bc0 + k0 + 16);
        const v16h b1 = load_frag(bc1 + k0, bc1 + k0 + 16);
        acc[0][0] = __builtin_amdgcn_wmma_f32_16x16x32_f16(false, a0, false, b0,
                                                           (short)0, acc[0][0], false, false);
        acc[0][1] = __builtin_amdgcn_wmma_f32_16x16x32_f16(false, a0, false, b1,
                                                           (short)0, acc[0][1], false, false);
        acc[1][0] = __builtin_amdgcn_wmma_f32_16x16x32_f16(false, a1, false, b0,
                                                           (short)0, acc[1][0], false, false);
        acc[1][1] = __builtin_amdgcn_wmma_f32_16x16x32_f16(false, a1, false, b1,
                                                           (short)0, acc[1][1], false, false);
    }

    // epilogue: bias + optional relu, f32 and/or f16 stores
#pragma unroll
    for (int t = 0; t < 2; ++t) {
        const int n = n0 + wn * 32 + t * 16 + mloc;
        const float bv = bias ? bias[n] : 0.f;
#pragma unroll
        for (int mt = 0; mt < 2; ++mt) {
#pragma unroll
            for (int i = 0; i < 8; ++i) {
                const int m = m0 + wm * 32 + mt * 16 + half * 8 + i;
                float v = acc[mt][t][i] + bv;
                if (relu) v = fmaxf(v, 0.f);
                if (Cf) Cf[(size_t)m * N + n] = v;
                if (Ch) {
                    size_t idx;
                    if (cht == 0) {
                        idx = (size_t)m * N + n;
                    } else {           // V: [B,L,(h,d)] -> [B,NH,DH,L], key-contiguous
                        const int bb = m >> 8, l = m & 255;
                        const int hh = n >> 5, d = n & 31;
                        idx = (((size_t)bb * NH_SZ + hh) * DH_SZ + d) * L_SZ + l;
                    }
                    Ch[idx] = (_Float16)v;
                }
            }
        }
    }
}

// ============================================================
// Kernel 3: fused attention. 128 threads = 4 waves per (b,h,64 query rows);
// each wave owns a 16-query tile. Q/K/V fragments: global b128 loads
// (V pre-transposed to [B,NH,DH,L]). LDS only holds f16 exp-scores.
// Softmax keeps UNNORMALIZED exp; 1/sum folded into output scaling.
// ============================================================
#define PSTR 264   // padded row stride in halves (528 B, 16B-aligned)

__global__ void __launch_bounds__(128)
attention_kernel(const _Float16* __restrict__ Q, const _Float16* __restrict__ K,
                 const _Float16* __restrict__ Vt, float* __restrict__ O)
{
    alignas(16) __shared__ _Float16 sP[4][16 * PSTR];        // 33 KB exp(scores)

    const int lane = threadIdx.x & 31;
    const int w    = threadIdx.x >> 5;
    const int mloc = lane & 15;
    const int half = lane >> 4;
    const int h = blockIdx.y;
    const int b = blockIdx.z;
    const int q0 = blockIdx.x * 64 + w * 16;
    const float scale = 0.17677669529663687f;   // 1/sqrt(DH)

    // Q fragment straight from global (two b128 loads)
    const _Float16* qr = Q + ((size_t)b * L_SZ + q0 + mloc) * H_SZ + h * DH_SZ;
    const v16h a = load_frag(qr + half * 8, qr + 16 + half * 8);

    // scores -> scaled -> f16 into per-wave sP region
    for (int j = 0; j < L_SZ / 16; ++j) {
        const _Float16* kr = K + ((size_t)b * L_SZ + j * 16 + mloc) * H_SZ + h * DH_SZ;
        const v16h bb = load_frag(kr + half * 8, kr + 16 + half * 8);
        v8f c = {};
        c = __builtin_amdgcn_wmma_f32_16x16x32_f16(false, a, false, bb,
                                                   (short)0, c, false, false);
#pragma unroll
        for (int i = 0; i < 8; ++i)
            sP[w][(half * 8 + i) * PSTR + j * 16 + mloc] = (_Float16)(c[i] * scale);
    }
    __syncthreads();

    // softmax (vectorized v8h LDS traffic); 2 lanes per row split the 256 cols
    float rsum;
    {
        const int base = mloc * PSTR + half * 128;
        float mx = -3.4e38f;
#pragma unroll
        for (int ch = 0; ch < 16; ++ch) {
            v8h cv = *(const v8h*)&sP[w][base + ch * 8];
#pragma unroll
            for (int e = 0; e < 8; ++e) mx = fmaxf(mx, (float)cv[e]);
        }
        mx = fmaxf(mx, __shfl_xor(mx, 16, 32));
        float sum = 0.f;
#pragma unroll
        for (int ch = 0; ch < 16; ++ch) {
            v8h cv = *(const v8h*)&sP[w][base + ch * 8];
            v8h ev;
#pragma unroll
            for (int e = 0; e < 8; ++e) {
                float x = __expf((float)cv[e] - mx);
                sum += x;
                ev[e] = (_Float16)x;
            }
            *(v8h*)&sP[w][base + ch * 8] = ev;
        }
        sum += __shfl_xor(sum, 16, 32);
        rsum = sum;                       // full row sum for row `mloc`
    }
    __syncthreads();

    // per-output-row normalization factors (C-layout row = i + 8*half)
    float invr[8];
#pragma unroll
    for (int i = 0; i < 8; ++i)
        invr[i] = 1.f / __shfl(rsum, i + 8 * half, 32);

    // out = exp(S) @ V, scaled by 1/rowsum; V fragments from global (key-contig)
    const _Float16* vbase = Vt + (((size_t)b * NH_SZ + h) * DH_SZ) * L_SZ;
#pragma unroll
    for (int nt = 0; nt < 2; ++nt) {
        v8f acc = {};
        const _Float16* vcol = vbase + (size_t)(nt * 16 + mloc) * L_SZ;
        for (int kt = 0; kt < L_SZ / 32; ++kt) {
            const _Float16* pr = &sP[w][mloc * PSTR + kt * 32];
            const v16h pa = load_frag(pr + half * 8, pr + 16 + half * 8);
            const v16h vb = load_frag(vcol + kt * 32 + half * 8,
                                      vcol + kt * 32 + 16 + half * 8);
            acc = __builtin_amdgcn_wmma_f32_16x16x32_f16(false, pa, false, vb,
                                                         (short)0, acc, false, false);
        }
#pragma unroll
        for (int i = 0; i < 8; ++i)
            O[((size_t)b * L_SZ + q0 + half * 8 + i) * H_SZ
              + h * DH_SZ + nt * 16 + mloc] = acc[i] * invr[i];
    }
}

// ============================================================
// Kernel 4: residual + LayerNorm (H=512), one wave per row.
// Writes f32 and (optionally) a f16 shadow for the next GEMM's A.
// ============================================================
__global__ void __launch_bounds__(256)
add_ln_kernel(const float* __restrict__ a, const float* __restrict__ x,
              const float* __restrict__ g, const float* __restrict__ beta,
              float* __restrict__ out, _Float16* __restrict__ outh)
{
    const int lane = threadIdx.x & 31;
    const int wave = threadIdx.x >> 5;
    const size_t row = (size_t)blockIdx.x * 8 + wave;
    const float* pa = a + row * H_SZ;
    const float* px = x + row * H_SZ;
    float vals[16];
    float s = 0.f;
#pragma unroll
    for (int i = 0; i < 16; ++i) {
        vals[i] = pa[lane + i * 32] + px[lane + i * 32];
        s += vals[i];
    }
#pragma unroll
    for (int off = 16; off > 0; off >>= 1) s += __shfl_xor(s, off, 32);
    const float mu = s * (1.f / H_SZ);
    float vs = 0.f;
#pragma unroll
    for (int i = 0; i < 16; ++i) {
        float d = vals[i] - mu;
        vs += d * d;
    }
#pragma unroll
    for (int off = 16; off > 0; off >>= 1) vs += __shfl_xor(vs, off, 32);
    const float inv = rsqrtf(vs * (1.f / H_SZ) + 1e-3f);
#pragma unroll
    for (int i = 0; i < 16; ++i) {
        int c = lane + i * 32;
        float v = (vals[i] - mu) * inv * g[c] + beta[c];
        out[row * H_SZ + c] = v;
        if (outh) outh[row * H_SZ + c] = (_Float16)v;
    }
}

// ============================================================
// Kernel 5: mask/transpose/sigmoid/entropy/bernoulli
// d_out = [samples(int32) | mask_scores(f32) | entropy(f32)], each [L,B,L]
// ============================================================
__global__ void __launch_bounds__(256)
final_out_kernel(const float* __restrict__ adj, float* __restrict__ out)
{
    const int idx = blockIdx.x * 256 + threadIdx.x;
    const int l2 = idx & 255;
    const int b  = (idx >> 8) & 63;
    const int l  = idx >> 14;
    float x = adj[((size_t)b * L_SZ + l) * L_SZ + l2];
    if (l == l2) x -= MASK_BIG;
    const float p  = 1.f / (1.f + __expf(-x));
    const float ax = fabsf(x);
    const float lse = log1pf(__expf(-ax));
    const float sp_pos = lse + fmaxf(x, 0.f);
    const float sp_neg = lse + fmaxf(-x, 0.f);
    const float ent = p * sp_neg + (1.f - p) * sp_pos;
    unsigned u = (unsigned)idx * 2654435761u + 0x9e3779b9u;
    u ^= u >> 16; u *= 2246822519u; u ^= u >> 13; u *= 3266489917u; u ^= u >> 16;
    const float r = (float)(u >> 8) * (1.f / 16777216.f);
    const int total = L_SZ * B_SZ * L_SZ;
    ((int*)out)[idx]     = (r < p) ? 1 : 0;
    out[total + idx]     = x;
    out[2 * total + idx] = ent;
}

// ============================================================
// Host launch
// ============================================================
extern "C" void kernel_launch(void* const* d_in, const int* in_sizes, int n_in,
                              void* d_out, int out_size, void* d_ws, size_t ws_size,
                              hipStream_t stream)
{
    (void)in_sizes; (void)n_in; (void)out_size; (void)ws_size;

    const float* inputs = (const float*)d_in[0];
    const float* emb_w  = (const float*)d_in[1];
    const float* emb_b  = (const float*)d_in[2];
    const float* out_w  = (const float*)d_in[3];
    const float* out_b  = (const float*)d_in[4];

    char* ws = (char*)d_ws;
    _Float16* xcat_h = (_Float16*)(ws + 0);             //  4 MB  [16384,128] f16
    float*    enc    = (float*)(ws + 4194304);          // 32 MB  [16384,512] f32
    _Float16* ench   = (_Float16*)(ws + 37748736);      // 16 MB  f16 shadow
    float*    tmp    = (float*)(ws + 54525952);         // 32 MB  attn-out / h2 / adj
    _Float16* tmph   = (_Float16*)(ws + 88080384);      // 16 MB  h1 f16
    _Float16* qh     = (_Float16*)(ws + 104857600);     // 16 MB
    _Float16* kh     = (_Float16*)(ws + 121634816);     // 16 MB
    _Float16* vt     = (_Float16*)(ws + 138412032);     // 16 MB  [B,NH,DH,L]
    _Float16* wc     = (_Float16*)(ws + 155189248);     // 0.5 MB weight scratch

    auto gemm = [&](const _Float16* A, const float* W, const float* bias,
                    float* Cf, _Float16* Ch, int M, int N, int K, int relu, int cht) {
        convert_w_kernel<<<(K * N) / 256, 256, 0, stream>>>(W, wc, K, N);
        dim3 grid(N / 128, M / 64);
        wmma_gemm_kernel<<<grid, 256, 0, stream>>>(A, wc, bias, Cf, Ch, M, N, K, relu, cht);
    };

    // embed
    mean_concat_kernel<<<B_SZ, 256, 0, stream>>>(inputs, xcat_h);
    gemm(xcat_h, emb_w, emb_b, enc, ench, M_ROWS, H_SZ, 2 * DIN_SZ, 1, 0);

    for (int s = 0; s < 3; ++s) {
        const float* wq    = (const float*)d_in[5 + 14 * s + 0];
        const float* bq    = (const float*)d_in[5 + 14 * s + 1];
        const float* wk    = (const float*)d_in[5 + 14 * s + 2];
        const float* bk    = (const float*)d_in[5 + 14 * s + 3];
        const float* wv    = (const float*)d_in[5 + 14 * s + 4];
        const float* bv    = (const float*)d_in[5 + 14 * s + 5];
        const float* ln1_g = (const float*)d_in[5 + 14 * s + 6];
        const float* ln1_b = (const float*)d_in[5 + 14 * s + 7];
        const float* w1    = (const float*)d_in[5 + 14 * s + 8];
        const float* b1    = (const float*)d_in[5 + 14 * s + 9];
        const float* w2    = (const float*)d_in[5 + 14 * s + 10];
        const float* b2    = (const float*)d_in[5 + 14 * s + 11];
        const float* ln2_g = (const float*)d_in[5 + 14 * s + 12];
        const float* ln2_b = (const float*)d_in[5 + 14 * s + 13];

        gemm(ench, wq, bq, nullptr, qh, M_ROWS, H_SZ, H_SZ, 1, 0);
        gemm(ench, wk, bk, nullptr, kh, M_ROWS, H_SZ, H_SZ, 1, 0);
        gemm(ench, wv, bv, nullptr, vt, M_ROWS, H_SZ, H_SZ, 1, 1);   // transposed V

        attention_kernel<<<dim3(L_SZ / 64, NH_SZ, B_SZ), 128, 0, stream>>>(qh, kh, vt, tmp);
        add_ln_kernel<<<M_ROWS / 8, 256, 0, stream>>>(tmp, enc, ln1_g, ln1_b, enc, ench);

        gemm(ench, w1, b1, nullptr, tmph, M_ROWS, H_SZ, H_SZ, 1, 0);   // h1 (f16 only)
        gemm(tmph, w2, b2, tmp, nullptr, M_ROWS, H_SZ, H_SZ, 1, 0);    // h2 (f32)
        add_ln_kernel<<<M_ROWS / 8, 256, 0, stream>>>(tmp, enc, ln2_g, ln2_b, enc, ench);
    }

    // output head: [16384,512] x [512,256]
    gemm(ench, out_w, out_b, tmp, nullptr, M_ROWS, L_SZ, H_SZ, 0, 0);
    final_out_kernel<<<(L_SZ * B_SZ * L_SZ) / 256, 256, 0, stream>>>(tmp, (float*)d_out);
}
